// PROTOTYPE_20753281975111
// MI455X (gfx1250) — matrix operand order repaired and checked by reading, not compiled, not run on hardware
//
#include <hip/hip_runtime.h>
#include <hip/hip_bf16.h>
#include <math.h>

#define NN     25600
#define SS     20
#define EE     64
#define DEG    50
#define GROUPN 50
#define KTOP   20
#define ITEMN  50000
#define NB     (NN / GROUPN)   /* 512 core users */

typedef __attribute__((ext_vector_type(16))) _Float16 v16h;
typedef __attribute__((ext_vector_type(8)))  _Float16 v8h;
typedef __attribute__((ext_vector_type(8)))  float    v8f;

static __device__ __forceinline__ v8f wmma16x16x32(v16h a, v16h b, v8f c) {
  // D = A(16x32 f16) * B(32x16 f16) + C(16x16 f32)
  return __builtin_amdgcn_wmma_f32_16x16x32_f16(false, a, false, b, (short)0, c,
                                                false, false);
}
static __device__ __forceinline__ float sigf(float x) {
  return 1.f / (1.f + __expf(-x));
}
static __device__ __forceinline__ float4 ldf4(const float* p) {
  return *(const float4*)p;
}
static __device__ __forceinline__ float4 sub4(float4 a, float4 b) {
  return make_float4(a.x - b.x, a.y - b.y, a.z - b.z, a.w - b.w);
}
// Build a v16h A/B fragment from four float4 runs (converted to f16).
static __device__ __forceinline__ v16h cvt16(float4 a, float4 b, float4 c,
                                             float4 d) {
  v16h v;
  v[0]  = (_Float16)a.x; v[1]  = (_Float16)a.y;
  v[2]  = (_Float16)a.z; v[3]  = (_Float16)a.w;
  v[4]  = (_Float16)b.x; v[5]  = (_Float16)b.y;
  v[6]  = (_Float16)b.z; v[7]  = (_Float16)b.w;
  v[8]  = (_Float16)c.x; v[9]  = (_Float16)c.y;
  v[10] = (_Float16)c.z; v[11] = (_Float16)c.w;
  v[12] = (_Float16)d.x; v[13] = (_Float16)d.y;
  v[14] = (_Float16)d.z; v[15] = (_Float16)d.w;
  return v;
}

// ---------------------------------------------------------------------------
// Kernel 1: item-emb gather + LSTM, record h at t == length-1.
// One wave = 16 nodes. Per step: g[16,256] = x@W_ih^T + h@W_hh^T as 64
// v_wmma_f32_16x16x32_f16, processed one column-block jt at a time so only
// 4 accumulator tiles (32 VGPRs) are live -> no scratch spills.
// W packed once into LDS in B-fragment order (2x ds_load_b128 per fragment).
// ---------------------------------------------------------------------------
__global__ void lstm_kernel(const int* __restrict__ session,
                            const int* __restrict__ length,
                            const float* __restrict__ item_emb,
                            const float* __restrict__ W_ih,
                            const float* __restrict__ W_hh,
                            const float* __restrict__ b_ih,
                            const float* __restrict__ b_hh,
                            float* __restrict__ rep0) {
  extern __shared__ char smem[];
  _Float16* wih   = (_Float16*)smem;   // 2 kc * 16 nt * 32 lanes * 16 = 16384
  _Float16* whh   = wih + 16384;
  _Float16* hbase = whh + 16384;       // per-wave 16x64 f16 hidden state

  // Pack W^T into B-fragment order: frag(kc,nt), lane l holds
  // K = kc*32 + (l>=16?8:0) + (i&7) + (i>=8?16:0), column n = nt*16 + (l&15); B[k][n] = W[n][k].
  for (int idx = threadIdx.x; idx < 16384; idx += blockDim.x) {
    int i  = idx & 15;
    int l  = (idx >> 4) & 31;
    int nt = (idx >> 9) & 15;
    int kc = idx >> 13;
    int k  = kc * 32 + ((l & 16) ? 8 : 0) + (i & 7) + ((i & 8) ? 16 : 0);
    int n  = nt * 16 + (l & 15);
    wih[idx] = (_Float16)W_ih[n * EE + k];
    whh[idx] = (_Float16)W_hh[n * EE + k];
  }

  int lane = threadIdx.x & 31;
  int wave = threadIdx.x >> 5;
  int base = (blockIdx.x * (blockDim.x >> 5) + wave) * 16;   // node tile base
  _Float16* hld = hbase + wave * (16 * EE);
  for (int i = lane; i < 16 * EE; i += 32) hld[i] = (_Float16)0.f;
  __syncthreads();

  int moff = (lane & 16) ? 8 : 0;   // C-layout row offset for this half-wave
  int ln   = lane & 15;

  float bias[16];
#pragma unroll
  for (int T = 0; T < 16; ++T) {
    int n = T * 16 + ln;
    bias[T] = b_ih[n] + b_hh[n];
  }
  int lenr[8];
#pragma unroll
  for (int r = 0; r < 8; ++r) lenr[r] = length[base + moff + r];

  float creg[4][8];
#pragma unroll
  for (int jt = 0; jt < 4; ++jt)
#pragma unroll
    for (int r = 0; r < 8; ++r) creg[jt][r] = 0.f;

  int arow = base + ln;   // the A-matrix row this lane supplies

  for (int t = 0; t < SS; ++t) {
    // ---- A fragments: x_t (item_emb, float4 loads) + h_{t-1} (LDS b128) ----
    int item = session[arow * SS + t];
    const float* xr = item_emb + (size_t)item * EE;
    const _Float16* hr = hld + ln * EE;
    v16h ax[2], ah[2];
#pragma unroll
    for (int kc = 0; kc < 2; ++kc) {
      int koff = kc * 32 + ((lane & 16) ? 8 : 0);
      ax[kc] = cvt16(ldf4(xr + koff), ldf4(xr + koff + 4),
                     ldf4(xr + koff + 16), ldf4(xr + koff + 20));
      v8h h0 = *(const v8h*)(hr + koff);
      v8h h1 = *(const v8h*)(hr + koff + 16);
      v16h hh;
#pragma unroll
      for (int i = 0; i < 8; ++i) { hh[i] = h0[i]; hh[8 + i] = h1[i]; }
      ah[kc] = hh;
    }
    __syncthreads();   // all lanes done reading hld before it is rewritten

    // ---- per column-block: 4 gate tiles (i,f,g,o), then gate math ----
#pragma unroll
    for (int jt = 0; jt < 4; ++jt) {
      v8f acc[4];
      v8f zero = {};
#pragma unroll
      for (int g = 0; g < 4; ++g) acc[g] = zero;
#pragma unroll
      for (int g = 0; g < 4; ++g) {
        int nt = g * 4 + jt;   // tile column block for gate g
        v16h b;
        b = *(const v16h*)(wih + ((0 * 16 + nt) * 32 + lane) * 16);
        acc[g] = wmma16x16x32(ax[0], b, acc[g]);
        b = *(const v16h*)(wih + ((1 * 16 + nt) * 32 + lane) * 16);
        acc[g] = wmma16x16x32(ax[1], b, acc[g]);
        b = *(const v16h*)(whh + ((0 * 16 + nt) * 32 + lane) * 16);
        acc[g] = wmma16x16x32(ah[0], b, acc[g]);
        b = *(const v16h*)(whh + ((1 * 16 + nt) * 32 + lane) * 16);
        acc[g] = wmma16x16x32(ah[1], b, acc[g]);
      }
#pragma unroll
      for (int r = 0; r < 8; ++r) {
        float iv = acc[0][r] + bias[jt];
        float fv = acc[1][r] + bias[4 + jt];
        float gv = acc[2][r] + bias[8 + jt];
        float ov = acc[3][r] + bias[12 + jt];
        float c  = sigf(fv) * creg[jt][r] + sigf(iv) * tanhf(gv);
        float h  = sigf(ov) * tanhf(c);
        creg[jt][r] = c;
        int m = moff + r;
        int j = jt * 16 + ln;
        hld[m * EE + j] = (_Float16)h;
        if (t == lenr[r] - 1) rep0[(size_t)(base + m) * EE + j] = h;
      }
    }
    __syncthreads();   // h writes visible before next step's fragment reads
  }
}

// ---------------------------------------------------------------------------
// Kernel 2: edge MLP. rows = (n,d) pairs; feat[128] = [src || src - self].
// 16 WMMAs per wave (4 kc x 4 nt), relu+dot(w2) via half-wave shfl reduce.
// ---------------------------------------------------------------------------
__global__ void edge_mlp_kernel(const float* __restrict__ rep0,
                                const int* __restrict__ src_ids,
                                const float* __restrict__ mlp_W1,
                                const float* __restrict__ mlp_b1,
                                const float* __restrict__ mlp_W2,
                                float* __restrict__ score) {
  __shared__ _Float16 w1p[4 * 4 * 32 * 16];   // 8192 halves = 16KB
  for (int idx = threadIdx.x; idx < 8192; idx += blockDim.x) {
    int i  = idx & 15;
    int l  = (idx >> 4) & 31;
    int nt = (idx >> 9) & 3;
    int kc = idx >> 11;
    int k  = kc * 32 + ((l & 16) ? 8 : 0) + (i & 7) + ((i & 8) ? 16 : 0);
    int n  = nt * 16 + (l & 15);
    w1p[idx] = (_Float16)mlp_W1[n * (2 * EE) + k];
  }
  __syncthreads();

  int lane = threadIdx.x & 31;
  int wave = threadIdx.x >> 5;
  int rowbase = (blockIdx.x * (blockDim.x >> 5) + wave) * 16;
  int ln   = lane & 15;
  int row  = rowbase + ln;
  int self = row / DEG;
  int src  = src_ids[row];
  const float* sr = rep0 + (size_t)src * EE;
  const float* nr = rep0 + (size_t)self * EE;

  v16h a[4];
#pragma unroll
  for (int kc = 0; kc < 4; ++kc) {
    int koff = kc * 32 + ((lane & 16) ? 8 : 0);
    if (kc < 2) {   // first 64 dims: src_rep
      a[kc] = cvt16(ldf4(sr + koff), ldf4(sr + koff + 4),
                    ldf4(sr + koff + 16), ldf4(sr + koff + 20));
    } else {        // last 64 dims: src_rep - self_rep
      int kb = koff - EE;
      a[kc] = cvt16(sub4(ldf4(sr + kb),      ldf4(nr + kb)),
                    sub4(ldf4(sr + kb + 4),  ldf4(nr + kb + 4)),
                    sub4(ldf4(sr + kb + 16), ldf4(nr + kb + 16)),
                    sub4(ldf4(sr + kb + 20), ldf4(nr + kb + 20)));
    }
  }

  v8f acc[4];
  v8f zero = {};
#pragma unroll
  for (int T = 0; T < 4; ++T) acc[T] = zero;
#pragma unroll
  for (int nt = 0; nt < 4; ++nt)
#pragma unroll
    for (int kc = 0; kc < 4; ++kc) {
      v16h b = *(const v16h*)(w1p + ((kc * 4 + nt) * 32 + lane) * 16);
      acc[nt] = wmma16x16x32(a[kc], b, acc[nt]);
    }

  float b1v[4], w2v[4];
#pragma unroll
  for (int T = 0; T < 4; ++T) {
    b1v[T] = mlp_b1[T * 16 + ln];
    w2v[T] = mlp_W2[T * 16 + ln];
  }
  int moff = (lane & 16) ? 8 : 0;
#pragma unroll
  for (int r = 0; r < 8; ++r) {
    float p = 0.f;
#pragma unroll
    for (int T = 0; T < 4; ++T) p += fmaxf(acc[T][r] + b1v[T], 0.f) * w2v[T];
    p += __shfl_xor(p, 1, 32);
    p += __shfl_xor(p, 2, 32);
    p += __shfl_xor(p, 4, 32);
    p += __shfl_xor(p, 8, 32);
    if (ln == 0) score[rowbase + moff + r] = p;
  }
}

// ---------------------------------------------------------------------------
// Kernel 3: friend rep = relu([user_emb[ids] || rep0] @ W1^T); core keeps rep0
// ---------------------------------------------------------------------------
__global__ void friend_rep_kernel(const float* __restrict__ rep0,
                                  const float* __restrict__ user_emb,
                                  const int* __restrict__ ids,
                                  const float* __restrict__ W1,
                                  float* __restrict__ urep) {
  __shared__ _Float16 w1p[4 * 4 * 32 * 16];
  for (int idx = threadIdx.x; idx < 8192; idx += blockDim.x) {
    int i  = idx & 15;
    int l  = (idx >> 4) & 31;
    int nt = (idx >> 9) & 3;
    int kc = idx >> 11;
    int k  = kc * 32 + ((l & 16) ? 8 : 0) + (i & 7) + ((i & 8) ? 16 : 0);
    int n  = nt * 16 + (l & 15);
    w1p[idx] = (_Float16)W1[n * (2 * EE) + k];
  }
  __syncthreads();

  int lane = threadIdx.x & 31;
  int wave = threadIdx.x >> 5;
  int rowbase = (blockIdx.x * (blockDim.x >> 5) + wave) * 16;
  int ln   = lane & 15;
  int node = rowbase + ln;
  const float* ur = user_emb + (size_t)ids[node] * EE;
  const float* hr = rep0 + (size_t)node * EE;

  v16h a[4];
#pragma unroll
  for (int kc = 0; kc < 4; ++kc) {
    int koff = kc * 32 + ((lane & 16) ? 8 : 0);
    const float* p = (kc < 2) ? (ur + koff) : (hr + (koff - EE));
    a[kc] = cvt16(ldf4(p), ldf4(p + 4), ldf4(p + 16), ldf4(p + 20));
  }

  v8f acc[4];
  v8f zero = {};
#pragma unroll
  for (int T = 0; T < 4; ++T) acc[T] = zero;
#pragma unroll
  for (int nt = 0; nt < 4; ++nt)
#pragma unroll
    for (int kc = 0; kc < 4; ++kc) {
      v16h b = *(const v16h*)(w1p + ((kc * 4 + nt) * 32 + lane) * 16);
      acc[nt] = wmma16x16x32(a[kc], b, acc[nt]);
    }

  int moff = (lane & 16) ? 8 : 0;
#pragma unroll
  for (int T = 0; T < 4; ++T)
#pragma unroll
    for (int r = 0; r < 8; ++r) {
      int m  = moff + r;
      int nd = rowbase + m;
      int j  = T * 16 + ln;
      float v = fmaxf(acc[T][r], 0.f);
      urep[(size_t)nd * EE + j] =
          (nd % GROUPN == 0) ? rep0[(size_t)nd * EE + j] : v;
    }
}

// ---------------------------------------------------------------------------
// Kernel 4: per-node top-k mask, pseudo-adjacency, coeff; core user outputs.
// ---------------------------------------------------------------------------
__global__ void topk_kernel(const float* __restrict__ score,
                            const int* __restrict__ src_ids,
                            const int* __restrict__ ids,
                            const int* __restrict__ keep_rate,
                            const float* __restrict__ init_score,
                            float* __restrict__ pseudo,
                            float* __restrict__ coeff,
                            float* __restrict__ out_sel,
                            float* __restrict__ out_ids) {
  int n = blockIdx.x * blockDim.x + threadIdx.x;
  if (n >= NN) return;
  float s[DEG];
  bool  msk[DEG];
  int   sel[KTOP];
  for (int d = 0; d < DEG; ++d) { s[d] = score[n * DEG + d]; msk[d] = false; }
  for (int k = 0; k < KTOP; ++k) {          // repeated strict argmax == top_k
    float best = -3.4e38f;
    int bi = 0;
    for (int d = 0; d < DEG; ++d)
      if (!msk[d] && s[d] > best) { best = s[d]; bi = d; }
    msk[bi] = true;
    sel[k]  = bi;
  }
  float kr   = (float)keep_rate[0];
  float csum = 0.f;
  for (int d = 0; d < DEG; ++d) {
    float x = msk[d] ? s[d] : (s[d] - 1e11f);
    float p = sigf(x);
    p = (1.f - kr) * p + kr * init_score[n * DEG + d];
    pseudo[n * DEG + d] = p;
    csum += p;
  }
  coeff[n] = csum + 1.f;
  if (n % GROUPN == 0) {
    int b = n / GROUPN;
    for (int k = 0; k < KTOP; ++k)
      out_sel[b * KTOP + k] = (float)ids[src_ids[n * DEG + sel[k]]];
    out_ids[b] = (float)ids[n];
  }
}

// ---------------------------------------------------------------------------
// Kernel 5: one aggregation layer: out = (sum_d pseudo*in[src] + in) / coeff.
// One wave per node, 2 dims per lane; 33MB of state lives in the 192MB L2.
// ---------------------------------------------------------------------------
__global__ void agg_kernel(const float* __restrict__ bin,
                           const float* __restrict__ pseudo,
                           const float* __restrict__ coeff,
                           const int* __restrict__ src_ids,
                           float* __restrict__ bout) {
  int lane = threadIdx.x & 31;
  int node = (blockIdx.x * blockDim.x + threadIdx.x) >> 5;
  if (node >= NN) return;
  float a0 = bin[(size_t)node * EE + lane];
  float a1 = bin[(size_t)node * EE + 32 + lane];
  for (int d = 0; d < DEG; ++d) {
    float w = pseudo[node * DEG + d];
    int   s = src_ids[node * DEG + d];
    a0 += w * bin[(size_t)s * EE + lane];
    a1 += w * bin[(size_t)s * EE + 32 + lane];
  }
  float ci = coeff[node];
  bout[(size_t)node * EE + lane]      = a0 / ci;
  bout[(size_t)node * EE + 32 + lane] = a1 / ci;
}

// ---------------------------------------------------------------------------
// Kernel 6: core readout logits: [512 x 64] x [64 x 50000] via WMMA tiles.
// One wave = one 16x16 output tile (2 k-chunks), all loads 128-bit.
// ---------------------------------------------------------------------------
__global__ void logits_kernel(const float* __restrict__ urep,
                              const float* __restrict__ buf,
                              const float* __restrict__ item_emb,
                              float* __restrict__ out) {
  int lane = threadIdx.x & 31;
  int wave = threadIdx.x >> 5;
  int gw   = blockIdx.x * (blockDim.x >> 5) + wave;
  const int ITILES = ITEMN / 16;   // 3125
  int rt = gw / ITILES;
  int it = gw % ITILES;
  int ln = lane & 15;
  int core = (rt * 16 + ln) * GROUPN;
  int col  = it * 16 + ln;
  const float* ar = urep + (size_t)core * EE;
  const float* br = buf + (size_t)core * EE;
  const float* ir = item_emb + (size_t)col * EE;

  v16h a[2], b[2];
#pragma unroll
  for (int kc = 0; kc < 2; ++kc) {
    int ka = kc * 32 + ((lane & 16) ? 8 : 0);
    a[kc] = cvt16(
        make_float4(ar[ka + 0] + br[ka + 0], ar[ka + 1] + br[ka + 1],
                    ar[ka + 2] + br[ka + 2], ar[ka + 3] + br[ka + 3]),
        make_float4(ar[ka + 4] + br[ka + 4], ar[ka + 5] + br[ka + 5],
                    ar[ka + 6] + br[ka + 6], ar[ka + 7] + br[ka + 7]),
        make_float4(ar[ka + 16] + br[ka + 16], ar[ka + 17] + br[ka + 17],
                    ar[ka + 18] + br[ka + 18], ar[ka + 19] + br[ka + 19]),
        make_float4(ar[ka + 20] + br[ka + 20], ar[ka + 21] + br[ka + 21],
                    ar[ka + 22] + br[ka + 22], ar[ka + 23] + br[ka + 23]));
    int kb = kc * 32 + ((lane & 16) ? 8 : 0);
    b[kc] = cvt16(ldf4(ir + kb), ldf4(ir + kb + 4), ldf4(ir + kb + 16),
                  ldf4(ir + kb + 20));
  }
  v8f acc = {};
  acc = wmma16x16x32(a[0], b[0], acc);
  acc = wmma16x16x32(a[1], b[1], acc);

  int moff = (lane & 16) ? 8 : 0;
#pragma unroll
  for (int r = 0; r < 8; ++r)
    out[(size_t)(rt * 16 + moff + r) * ITEMN + col] = acc[r];
}

// ---------------------------------------------------------------------------
// Kernel 7: in-place row softmax over 50000 items; one block per row.
// ---------------------------------------------------------------------------
__global__ void softmax_kernel(float* __restrict__ out) {
  __shared__ float red[256];
  int row = blockIdx.x;
  float* p = out + (size_t)row * ITEMN;
  int tid = threadIdx.x;

  float mx = -3.4e38f;
  for (int i = tid; i < ITEMN; i += blockDim.x) mx = fmaxf(mx, p[i]);
  red[tid] = mx;
  __syncthreads();
  for (int s = 128; s > 0; s >>= 1) {
    if (tid < s) red[tid] = fmaxf(red[tid], red[tid + s]);
    __syncthreads();
  }
  float rmax = red[0];
  __syncthreads();

  float sum = 0.f;
  for (int i = tid; i < ITEMN; i += blockDim.x) sum += __expf(p[i] - rmax);
  red[tid] = sum;
  __syncthreads();
  for (int s = 128; s > 0; s >>= 1) {
    if (tid < s) red[tid] += red[tid + s];
    __syncthreads();
  }
  float inv = 1.f / red[0];
  __syncthreads();

  for (int i = tid; i < ITEMN; i += blockDim.x) p[i] = __expf(p[i] - rmax) * inv;
}

// ---------------------------------------------------------------------------
extern "C" void kernel_launch(void* const* d_in, const int* in_sizes, int n_in,
                              void* d_out, int out_size, void* d_ws,
                              size_t ws_size, hipStream_t stream) {
  (void)in_sizes; (void)n_in; (void)out_size; (void)ws_size;

  const int*   session   = (const int*)d_in[0];
  const int*   length    = (const int*)d_in[1];
  const int*   src_ids   = (const int*)d_in[2];
  const int*   ids       = (const int*)d_in[3];
  /* d_in[4] = group_size (compile-time GROUPN) */
  const int*   keep_rate = (const int*)d_in[5];
  const float* initsc    = (const float*)d_in[6];
  const float* item_emb  = (const float*)d_in[7];
  const float* user_emb  = (const float*)d_in[8];
  const float* W_ih      = (const float*)d_in[9];
  const float* W_hh      = (const float*)d_in[10];
  const float* b_ih      = (const float*)d_in[11];
  const float* b_hh      = (const float*)d_in[12];
  const float* mlp_W1    = (const float*)d_in[13];
  const float* mlp_b1    = (const float*)d_in[14];
  const float* mlp_W2    = (const float*)d_in[15];
  const float* W1        = (const float*)d_in[16];
  float* out = (float*)d_out;

  float* ws     = (float*)d_ws;
  float* rep0   = ws;
  float* urep   = rep0 + (size_t)NN * EE;
  float* bufA   = urep + (size_t)NN * EE;
  float* bufB   = bufA + (size_t)NN * EE;
  float* score  = bufB + (size_t)NN * EE;
  float* pseudo = score + (size_t)NN * DEG;
  float* coeff  = pseudo + (size_t)NN * DEG;

  float* out_sel = out + (size_t)NB * ITEMN;
  float* out_ids = out_sel + (size_t)NB * KTOP;

  // 1) LSTM: 1600 waves of 16 nodes, 8 waves/block; dynamic LDS:
  //    64KB packed fp16 weights + 8 * 2KB hidden state = 80KB (< 320KB/WGP).
  {
    size_t shmem = (size_t)(2 * 16384 + 8 * 16 * EE) * 2;   // 81920 bytes
    lstm_kernel<<<dim3(NN / 16 / 8), dim3(256), shmem, stream>>>(
        session, length, item_emb, W_ih, W_hh, b_ih, b_hh, rep0);
  }
  // 2) edge MLP: 1.28M rows -> 80000 waves
  edge_mlp_kernel<<<dim3(NN * DEG / 16 / 8), dim3(256), 0, stream>>>(
      rep0, src_ids, mlp_W1, mlp_b1, mlp_W2, score);
  // 3) friend projection: 1600 waves
  friend_rep_kernel<<<dim3(NN / 16 / 8), dim3(256), 0, stream>>>(
      rep0, user_emb, ids, W1, urep);
  // 4) top-k / pseudo adjacency / coeff + int outputs
  topk_kernel<<<dim3((NN + 255) / 256), dim3(256), 0, stream>>>(
      score, src_ids, ids, keep_rate, initsc, pseudo, coeff, out_sel, out_ids);
  // 5) two aggregation layers (ping-pong)
  agg_kernel<<<dim3(NN * 32 / 256), dim3(256), 0, stream>>>(
      urep, pseudo, coeff, src_ids, bufA);
  agg_kernel<<<dim3(NN * 32 / 256), dim3(256), 0, stream>>>(
      bufA, pseudo, coeff, src_ids, bufB);
  // 6) readout logits: 32 row-tiles * 3125 item-tiles = 100000 waves
  logits_kernel<<<dim3((NB / 16) * (ITEMN / 16) / 8), dim3(256), 0, stream>>>(
      urep, bufB, item_emb, out);
  // 7) in-place softmax, one block per core user
  softmax_kernel<<<dim3(NB), dim3(256), 0, stream>>>(out);
}
